// GAT_40802189312199
// MI455X (gfx1250) — matrix operand order repaired and checked by reading, not compiled, not run on hardware
//
#include <hip/hip_runtime.h>
#include <hip/hip_bf16.h>

#define NNODES 50000
#define NEDGES 800000
#define ETOT   (NEDGES + NNODES)
#define FIN    128
#define HEADS  4
#define HDIM   64
#define HC     (HEADS * HDIM)   // 256
#define NGRAPH 64
#define OUTC   10

typedef __attribute__((ext_vector_type(16))) __bf16 v16bf;
typedef __attribute__((ext_vector_type(8)))  float  v8f;

union FragBF { v16bf v; uint4 q[2]; unsigned u32[8]; unsigned short u16[16]; };

// ---- helpers -------------------------------------------------------------
__device__ __forceinline__ unsigned fenc(float f) {
  unsigned u = __float_as_uint(f);
  return (u & 0x80000000u) ? ~u : (u | 0x80000000u);   // monotonic float->uint
}
__device__ __forceinline__ float fdec(unsigned u) {
  return (u & 0x80000000u) ? __uint_as_float(u & 0x7FFFFFFFu)
                           : __uint_as_float(~u);
}
__device__ __forceinline__ unsigned short f2bf(float f) {
  unsigned u = __float_as_uint(f);
  u += 0x7FFFu + ((u >> 16) & 1u);                     // RNE
  return (unsigned short)(u >> 16);
}

// ---- convert f32 -> bf16 (for the large activation matrices) ------------
__global__ __launch_bounds__(256)
void cvt_f32_bf16(const float* __restrict__ s, unsigned short* __restrict__ d, long n) {
  long i = (long)blockIdx.x * blockDim.x + threadIdx.x;
  if (i < n) d[i] = f2bf(s[i]);
}

// ---- pack f32 weights into WMMA B-fragment layout -----------------------
// Bp dword index i = (((j * ktiles) + t) * 32 + lane) * 8 + v
//   j = n-tile, t = k-tile, element e of lane <-> k = t*32 + 16*(lane>>4)+e,
//   n = j*16 + (lane&15); dword v packs e=2v (lo), 2v+1 (hi).
__global__ __launch_bounds__(256)
void pack_b_frags(const float* __restrict__ B, unsigned* __restrict__ Bp,
                  int K, int Nc) {
  int total = (K >> 5) * (Nc >> 4) * 256;
  int i = blockIdx.x * blockDim.x + threadIdx.x;
  if (i >= total) return;
  int v    = i & 7;
  int lane = (i >> 3) & 31;
  int tile = i >> 8;
  int ktiles = K >> 5;
  int j = tile / ktiles;
  int t = tile - j * ktiles;
  int n  = (j << 4) + (lane & 15);
  int k0 = (t << 5) + ((lane >> 4) << 4) + (v << 1);
  unsigned lo = f2bf(B[(size_t)k0 * Nc + n]);
  unsigned hi = f2bf(B[(size_t)(k0 + 1) * Nc + n]);
  Bp[i] = lo | (hi << 16);
}

// ---- bf16 WMMA GEMM: C[M,Nc] = A[M,K] * B[K,Nc] -------------------------
// one wave per 16x64 output strip: 4 accumulators share one A fragment,
// 4 back-to-back v_wmma_f32_16x16x32_bf16 per K-step.
__global__ __launch_bounds__(256)
void wmma_gemm_bf16(const unsigned short* __restrict__ A,
                    const unsigned* __restrict__ Bp,
                    float* __restrict__ C,
                    int M, int Nc, int K) {
  int lane = threadIdx.x & 31;
  int wave = (int)((blockIdx.x * blockDim.x + threadIdx.x) >> 5);
  int tilesN4 = Nc >> 6;                            // groups of 4 n-tiles
  int nWaves = (M >> 4) * tilesN4;
  if (wave >= nWaves) return;                       // wave-uniform exit
  int tm = wave / tilesN4;
  int tg = wave - tm * tilesN4;
  int ktiles = K >> 5;
  int mrow = (tm << 4) + (lane & 15);
  int half = lane >> 4;
  const unsigned short* Arow = A + (size_t)mrow * K;

  v8f acc0 = {}, acc1 = {}, acc2 = {}, acc3 = {};
  for (int t = 0; t < ktiles; ++t) {
    FragBF a;
    // A 16x32 bf16 fragment = two contiguous 16B runs per lane
    a.q[0] = *(const uint4*)(Arow + (t << 5) + (half << 4));
    a.q[1] = *(const uint4*)(Arow + (t << 5) + 8 + (half << 4));
    FragBF b0, b1, b2, b3;
    {
      size_t base0 = ((size_t)(((tg << 2) + 0) * ktiles + t) * 32 + lane) << 3;
      size_t base1 = ((size_t)(((tg << 2) + 1) * ktiles + t) * 32 + lane) << 3;
      size_t base2 = ((size_t)(((tg << 2) + 2) * ktiles + t) * 32 + lane) << 3;
      size_t base3 = ((size_t)(((tg << 2) + 3) * ktiles + t) * 32 + lane) << 3;
      b0.q[0] = *(const uint4*)(Bp + base0); b0.q[1] = *(const uint4*)(Bp + base0 + 4);
      b1.q[0] = *(const uint4*)(Bp + base1); b1.q[1] = *(const uint4*)(Bp + base1 + 4);
      b2.q[0] = *(const uint4*)(Bp + base2); b2.q[1] = *(const uint4*)(Bp + base2 + 4);
      b3.q[0] = *(const uint4*)(Bp + base3); b3.q[1] = *(const uint4*)(Bp + base3 + 4);
    }
    acc0 = __builtin_amdgcn_wmma_f32_16x16x32_bf16(false, a.v, false, b0.v, (short)0, acc0, false, false);
    acc1 = __builtin_amdgcn_wmma_f32_16x16x32_bf16(false, a.v, false, b1.v, (short)0, acc1, false, false);
    acc2 = __builtin_amdgcn_wmma_f32_16x16x32_bf16(false, a.v, false, b2.v, (short)0, acc2, false, false);
    acc3 = __builtin_amdgcn_wmma_f32_16x16x32_bf16(false, a.v, false, b3.v, (short)0, acc3, false, false);
  }
  // store: C/D layout -> row = tm*16 + r + 8*half, col = tile_n*16 + (lane&15)
  int rowBase = (tm << 4) + (half << 3);
  int colBase = (tg << 6) + (lane & 15);
#pragma unroll
  for (int r = 0; r < 8; ++r) {
    float* crow = C + (size_t)(rowBase + r) * Nc + colBase;
    crow[0]  = acc0[r];
    crow[16] = acc1[r];
    crow[32] = acc2[r];
    crow[48] = acc3[r];
  }
}

// ---- per-(node,head) attention scores -----------------------------------
__global__ __launch_bounds__(256)
void scores_kernel(const float* __restrict__ xp, const float* __restrict__ av,
                   const float* __restrict__ bv, float* __restrict__ s,
                   float* __restrict__ d, int N, int H) {
  int i = blockIdx.x * blockDim.x + threadIdx.x;
  if (i >= N * H) return;
  int n = i / H, h = i - n * H;
  const float* row = xp + (size_t)n * (H * 64) + h * 64;
  const float* a = av + h * 64;
  const float* b = bv + h * 64;
  float ss = 0.f, dd = 0.f;
#pragma unroll 8
  for (int c = 0; c < 64; ++c) { float v = row[c]; ss += v * a[c]; dd += v * b[c]; }
  s[i] = ss; d[i] = dd;
}

// ---- edge pass 1: leaky-relu score, segment max over dst ----------------
__global__ __launch_bounds__(256)
void edge_max(const long long* __restrict__ ei, const float* __restrict__ s,
              const float* __restrict__ d, unsigned* __restrict__ m, int H) {
  int t = blockIdx.x * blockDim.x + threadIdx.x;
  if (t >= ETOT * H) return;
  int e = t / H, h = t - e * H;
  int src, dst;
  if (e < NEDGES) { src = (int)ei[e]; dst = (int)ei[NEDGES + e]; }
  else            { src = dst = e - NEDGES; }
  float x = s[src * H + h] + d[dst * H + h];
  x = (x > 0.f) ? x : 0.2f * x;                   // leaky_relu, slope 0.2
  atomicMax(m + dst * H + h, fenc(x));
}

// ---- edge pass 2: exp(e - max), segment sum denominator -----------------
__global__ __launch_bounds__(256)
void edge_exp(const long long* __restrict__ ei, const float* __restrict__ s,
              const float* __restrict__ d, const unsigned* __restrict__ m,
              float* __restrict__ ex, float* __restrict__ den, int H) {
  int t = blockIdx.x * blockDim.x + threadIdx.x;
  if (t >= ETOT * H) return;
  int e = t / H, h = t - e * H;
  int src, dst;
  if (e < NEDGES) { src = (int)ei[e]; dst = (int)ei[NEDGES + e]; }
  else            { src = dst = e - NEDGES; }
  float x = s[src * H + h] + d[dst * H + h];
  x = (x > 0.f) ? x : 0.2f * x;
  float v = __expf(x - fdec(m[dst * H + h]));
  ex[t] = v;
  atomicAdd(den + dst * H + h, v);
}

// ---- layer1 aggregation: wave per edge, 256 channels (4 heads x 64) -----
__global__ __launch_bounds__(256)
void aggregate1(const long long* __restrict__ ei, const float* __restrict__ xp,
                const float* __restrict__ ex, const float* __restrict__ den,
                float* __restrict__ agg) {
  int wave = (int)((blockIdx.x * blockDim.x + threadIdx.x) >> 5);
  int lane = threadIdx.x & 31;
  if (wave >= ETOT) return;
  int e = wave;
  int src, dst;
  if (e < NEDGES) { src = (int)ei[e]; dst = (int)ei[NEDGES + e]; }
  else            { src = dst = e - NEDGES; }
  // prefetch next edge's source row (global_prefetch_b8)
  if (e + 1 < ETOT) {
    int ns = (e + 1 < NEDGES) ? (int)ei[e + 1] : (e + 1 - NEDGES);
    __builtin_prefetch(xp + (size_t)ns * HC + lane * 8, 0, 1);
  }
  int h = lane >> 3;                               // 8 lanes per head
  float alpha = ex[(size_t)e * HEADS + h] / den[dst * HEADS + h];
  const float4* sr = (const float4*)(xp + (size_t)src * HC) + lane * 2;
  float4 v0 = sr[0], v1 = sr[1];
  float* out = agg + (size_t)dst * HC + lane * 8;
  atomicAdd(out + 0, alpha * v0.x); atomicAdd(out + 1, alpha * v0.y);
  atomicAdd(out + 2, alpha * v0.z); atomicAdd(out + 3, alpha * v0.w);
  atomicAdd(out + 4, alpha * v1.x); atomicAdd(out + 5, alpha * v1.y);
  atomicAdd(out + 6, alpha * v1.z); atomicAdd(out + 7, alpha * v1.w);
}

// ---- layer2 aggregation: wave per edge, 64 channels, single head --------
__global__ __launch_bounds__(256)
void aggregate2(const long long* __restrict__ ei, const float* __restrict__ xp,
                const float* __restrict__ ex, const float* __restrict__ den,
                float* __restrict__ agg) {
  int wave = (int)((blockIdx.x * blockDim.x + threadIdx.x) >> 5);
  int lane = threadIdx.x & 31;
  if (wave >= ETOT) return;
  int e = wave;
  int src, dst;
  if (e < NEDGES) { src = (int)ei[e]; dst = (int)ei[NEDGES + e]; }
  else            { src = dst = e - NEDGES; }
  float alpha = ex[e] / den[dst];
  float2 v = ((const float2*)(xp + (size_t)src * 64))[lane];
  float* out = agg + (size_t)dst * 64 + lane * 2;
  atomicAdd(out + 0, alpha * v.x);
  atomicAdd(out + 1, alpha * v.y);
}

// ---- bias + ELU (in place), optional bf16 mirror for next GEMM ----------
__global__ __launch_bounds__(256)
void bias_elu(float* __restrict__ a, const float* __restrict__ bias,
              unsigned short* __restrict__ obf, long N, int F) {
  long i = (long)blockIdx.x * blockDim.x + threadIdx.x;
  if (i >= N * F) return;
  int c = (int)(i % F);
  float v = a[i] + bias[c];
  v = (v > 0.f) ? v : (__expf(v) - 1.f);           // ELU
  a[i] = v;
  if (obf) obf[i] = f2bf(v);
}

// ---- global mean pool (sums + counts) -----------------------------------
__global__ __launch_bounds__(256)
void pool_kernel(const float* __restrict__ h2, const long long* __restrict__ batch,
                 float* __restrict__ pooled, float* __restrict__ cnt) {
  long i = (long)blockIdx.x * blockDim.x + threadIdx.x;
  if (i >= (long)NNODES * 64) return;
  int n = (int)(i >> 6), c = (int)(i & 63);
  int g = (int)batch[n];
  atomicAdd(pooled + g * 64 + c, h2[i]);
  if (c == 0) atomicAdd(cnt + g, 1.f);
}

// ---- final linear: out[G,10] = (pooled/cnt) @ Wl + bl -------------------
__global__ void final_lin(const float* __restrict__ pooled, const float* __restrict__ cnt,
                          const float* __restrict__ Wl, const float* __restrict__ bl,
                          float* __restrict__ out) {
  int i = threadIdx.x;
  if (i >= NGRAPH * OUTC) return;
  int g = i / OUTC, o = i - g * OUTC;
  float inv = 1.f / fmaxf(cnt[g], 1.f);
  float acc = bl[o];
#pragma unroll 8
  for (int c = 0; c < 64; ++c) acc += pooled[g * 64 + c] * inv * Wl[c * OUTC + o];
  out[i] = acc;
}

// =========================================================================
extern "C" void kernel_launch(void* const* d_in, const int* in_sizes, int n_in,
                              void* d_out, int out_size, void* d_ws, size_t ws_size,
                              hipStream_t stream) {
  const float*     x     = (const float*)d_in[0];
  const long long* ei    = (const long long*)d_in[1];
  const long long* batch = (const long long*)d_in[2];
  const float*     W1    = (const float*)d_in[3];
  const float*     a_s1  = (const float*)d_in[4];
  const float*     a_d1  = (const float*)d_in[5];
  const float*     b1    = (const float*)d_in[6];
  const float*     W2    = (const float*)d_in[7];
  const float*     a_s2  = (const float*)d_in[8];
  const float*     a_d2  = (const float*)d_in[9];
  const float*     b2    = (const float*)d_in[10];
  const float*     Wl    = (const float*)d_in[11];
  const float*     bl    = (const float*)d_in[12];
  float* out = (float*)d_out;

  // ---- workspace carve-out ----
  char* ws = (char*)d_ws;
  size_t off = 0;
  auto alloc = [&](size_t bytes) -> void* {
    void* p = ws + off;
    off = (off + bytes + 255) & ~(size_t)255;
    return p;
  };
  unsigned short* xbf  = (unsigned short*)alloc((size_t)NNODES * FIN * 2);
  unsigned* w1p = (unsigned*)alloc((size_t)(FIN / 32) * (HC / 16) * 256 * 4);
  unsigned* w2p = (unsigned*)alloc((size_t)(HC / 32) * (HDIM / 16) * 256 * 4);
  float*    xp1  = (float*)alloc((size_t)NNODES * HC * 4);
  float*    s1   = (float*)alloc((size_t)NNODES * HEADS * 4);
  float*    d1   = (float*)alloc((size_t)NNODES * HEADS * 4);
  unsigned* m1   = (unsigned*)alloc((size_t)NNODES * HEADS * 4);
  float*    den1 = (float*)alloc((size_t)NNODES * HEADS * 4);
  float*    ex1  = (float*)alloc((size_t)ETOT * HEADS * 4);
  float*    agg1 = (float*)alloc((size_t)NNODES * HC * 4);   // becomes h1
  unsigned short* h1bf = (unsigned short*)alloc((size_t)NNODES * HC * 2);
  float*    xp2  = (float*)alloc((size_t)NNODES * HDIM * 4);
  float*    s2   = (float*)alloc((size_t)NNODES * 4);
  float*    d2   = (float*)alloc((size_t)NNODES * 4);
  unsigned* m2   = (unsigned*)alloc((size_t)NNODES * 4);
  float*    den2 = (float*)alloc((size_t)NNODES * 4);
  float*    ex2  = (float*)alloc((size_t)ETOT * 4);
  float*    agg2 = (float*)alloc((size_t)NNODES * HDIM * 4); // becomes h2
  float*    pooled = (float*)alloc((size_t)NGRAPH * HDIM * 4);
  float*    cnt  = (float*)alloc((size_t)NGRAPH * 4);

  // ---- zero accumulators / segment-max buffers every launch ----
  hipMemsetAsync(m1,   0, (size_t)NNODES * HEADS * 4, stream);
  hipMemsetAsync(den1, 0, (size_t)NNODES * HEADS * 4, stream);
  hipMemsetAsync(agg1, 0, (size_t)NNODES * HC * 4, stream);
  hipMemsetAsync(m2,   0, (size_t)NNODES * 4, stream);
  hipMemsetAsync(den2, 0, (size_t)NNODES * 4, stream);
  hipMemsetAsync(agg2, 0, (size_t)NNODES * HDIM * 4, stream);
  hipMemsetAsync(pooled, 0, (size_t)NGRAPH * HDIM * 4, stream);
  hipMemsetAsync(cnt,  0, (size_t)NGRAPH * 4, stream);

  const int B = 256;
  auto nb = [](long n, int b) { return (int)((n + b - 1) / b); };

  // ---- layer 1 ----
  cvt_f32_bf16<<<nb((long)NNODES * FIN, B), B, 0, stream>>>(x, xbf, (long)NNODES * FIN);
  pack_b_frags<<<nb((long)(FIN / 32) * (HC / 16) * 256, B), B, 0, stream>>>(W1, w1p, FIN, HC);
  {
    int waves = (NNODES / 16) * (HC / 64);             // 12500 strips
    wmma_gemm_bf16<<<nb((long)waves * 32, B), B, 0, stream>>>(
        xbf, w1p, xp1, NNODES, HC, FIN);
  }
  scores_kernel<<<nb((long)NNODES * HEADS, B), B, 0, stream>>>(xp1, a_s1, a_d1, s1, d1, NNODES, HEADS);
  edge_max<<<nb((long)ETOT * HEADS, B), B, 0, stream>>>(ei, s1, d1, m1, HEADS);
  edge_exp<<<nb((long)ETOT * HEADS, B), B, 0, stream>>>(ei, s1, d1, m1, ex1, den1, HEADS);
  aggregate1<<<nb((long)ETOT * 32, B), B, 0, stream>>>(ei, xp1, ex1, den1, agg1);
  bias_elu<<<nb((long)NNODES * HC, B), B, 0, stream>>>(agg1, b1, h1bf, NNODES, HC);

  // ---- layer 2 ----
  pack_b_frags<<<nb((long)(HC / 32) * (HDIM / 16) * 256, B), B, 0, stream>>>(W2, w2p, HC, HDIM);
  {
    int waves = (NNODES / 16) * (HDIM / 64);           // 3125 strips
    wmma_gemm_bf16<<<nb((long)waves * 32, B), B, 0, stream>>>(
        h1bf, w2p, xp2, NNODES, HDIM, HC);
  }
  scores_kernel<<<nb((long)NNODES, B), B, 0, stream>>>(xp2, a_s2, a_d2, s2, d2, NNODES, 1);
  edge_max<<<nb((long)ETOT, B), B, 0, stream>>>(ei, s2, d2, m2, 1);
  edge_exp<<<nb((long)ETOT, B), B, 0, stream>>>(ei, s2, d2, m2, ex2, den2, 1);
  aggregate2<<<nb((long)ETOT * 32, B), B, 0, stream>>>(ei, xp2, ex2, den2, agg2);
  bias_elu<<<nb((long)NNODES * HDIM, B), B, 0, stream>>>(agg2, b2, (unsigned short*)nullptr, NNODES, HDIM);

  // ---- pool + final linear ----
  pool_kernel<<<nb((long)NNODES * HDIM, B), B, 0, stream>>>(agg2, batch, pooled, cnt);
  final_lin<<<1, NGRAPH * OUTC, 0, stream>>>(pooled, cnt, Wl, bl, out);
  (void)in_sizes; (void)n_in; (void)out_size; (void)ws_size;
}
